// ISA_32066225832608
// MI455X (gfx1250) — matrix operand order repaired and checked by reading, not compiled, not run on hardware
//
#include <hip/hip_runtime.h>
#include <hip/hip_bf16.h>
#include <cstdint>

// ---------------------------------------------------------------------------
// MI455X (gfx1250) slot-attention forward.
// - All large GEMMs: bf16 WMMA kernel (v_wmma_f32_16x16x32_bf16), 128x128
//   block tile, 8 wave32 waves x (2x4) 16x16 accumulators, f32 accumulate.
// - Both A and W tiles staged with CDNA5 async DMA
//   (global_load_async_to_lds_b128), double-buffered in LDS and pipelined
//   with s_wait_asynccnt 4 (in-order per-wave completion) for copy/compute
//   overlap. Weights are pre-transposed to [N][K] so the B tile is a plain
//   row-major async copy (no element-wise transpose in the hot loop).
// - bf16 storage for activations/weights halves HBM traffic (23.3 TB/s).
// - Small per-slot ops (LN, softmax-over-S, GRU, slot MLP: 32 rows) are
//   plain f32 VALU kernels.
// ---------------------------------------------------------------------------

#define BB   4
#define NPTS 4096
#define DIN  768
#define DD   256
#define NS   8
#define NITR 3

typedef __attribute__((ext_vector_type(16))) __bf16 v16bf;
typedef __attribute__((ext_vector_type(8)))  float  v8f;

__device__ __forceinline__ float blk_reduce256(float v, float* red) {
    int tid = threadIdx.x;
    red[tid] = v; __syncthreads();
    for (int off = 128; off > 0; off >>= 1) {
        if (tid < off) red[tid] += red[tid + off];
        __syncthreads();
    }
    float r = red[0];
    __syncthreads();
    return r;
}

// 16-byte global -> LDS async copy (ASYNCcnt tracked, GV addressing).
__device__ __forceinline__ void async_cp16(const void* g, void* l) {
    unsigned ldsa = (unsigned)(uintptr_t)l;                 // LDS byte address
    unsigned long long ga = (unsigned long long)(uintptr_t)g;
    asm volatile("global_load_async_to_lds_b128 %0, %1, off"
                 :: "v"(ldsa), "v"(ga) : "memory");
}

// ---------------------------------------------------------------------------
// f32 [K][N] -> bf16 transposed [N][K] (weights, one-time per launch)
// ---------------------------------------------------------------------------
__global__ void k_f2bt(const float* __restrict__ src, __bf16* __restrict__ dst,
                       int K, int N) {
    int idx = blockIdx.x * 256 + threadIdx.x;
    if (idx < K * N) {
        int k = idx / N, n = idx % N;
        dst[(size_t)n * K + k] = (__bf16)src[idx];
    }
}

// ---------------------------------------------------------------------------
// Row LayerNorm: f32 in -> bf16 out. One block (256 thr) per row of length R.
// ---------------------------------------------------------------------------
__global__ void k_ln(const float* __restrict__ x, const float* __restrict__ g,
                     const float* __restrict__ b, __bf16* __restrict__ ob, int R) {
    __shared__ float red[256];
    int row = blockIdx.x, tid = threadIdx.x;
    const float* xr = x + (size_t)row * R;
    float s1 = 0.f, s2 = 0.f;
    for (int i = tid; i < R; i += 256) { float v = xr[i]; s1 += v; s2 += v * v; }
    float mean = blk_reduce256(s1, red) / (float)R;
    float var  = blk_reduce256(s2, red) / (float)R - mean * mean;
    float inv  = rsqrtf(var + 1e-5f);
    for (int i = tid; i < R; i += 256)
        ob[(size_t)row * R + i] = (__bf16)((xr[i] - mean) * inv * g[i] + b[i]);
}

// ---------------------------------------------------------------------------
// WMMA GEMM: C[MxN] = act(A[MxK] @ W[KxN] + bias), A row-major bf16,
// Wt = W transposed ([N][K] bf16). Double-buffered async-DMA staging.
// ---------------------------------------------------------------------------
template<bool RELU, bool HASB, bool OUTB>
__global__ __launch_bounds__(256) void k_gemm(
    const __bf16* __restrict__ A, const __bf16* __restrict__ Wt,
    const float* __restrict__ bias, void* __restrict__ Cout,
    int M, int N, int K)
{
    __shared__ __bf16 sA[2][128][32];   // A tile, row-major       (2 x 8KB)
    __shared__ __bf16 sB[2][128][32];   // Wt tile: sB[n][k]       (2 x 8KB)
    const int tid  = threadIdx.x;
    const int lane = tid & 31, wave = tid >> 5;
    const int wm   = (wave & 3) * 32;
    const int wn   = (wave >> 2) * 64;
    const size_t bm = (size_t)blockIdx.x * 128;
    const size_t bn = (size_t)blockIdx.y * 128;

    // Per-thread chunk coordinates (2 x 16B chunks for each tile).
    const int r0 = tid >> 1;                   // chunk set 0: rows 0..127
    const int c0 = (tid & 1) << 4;             // cols 0 / 16   (wait: 32 cols)
    // 512 chunks: idx = tid + c*256 ; row = idx>>2 ; col = (idx&3)*8
    v8f acc[2][4] = {};

    auto issue_tile = [&](int buf, int k0) {
        #pragma unroll
        for (int c = 0; c < 2; ++c) {
            int idx = tid + c * 256;           // 0..511
            int r   = idx >> 2;
            int col = (idx & 3) << 3;
            async_cp16(A + (bm + r) * (size_t)K + (k0 + col), &sA[buf][r][col]);
        }
        #pragma unroll
        for (int c = 0; c < 2; ++c) {
            int idx = tid + c * 256;           // 0..511
            int n   = idx >> 2;
            int col = (idx & 3) << 3;
            async_cp16(Wt + (bn + n) * (size_t)K + (k0 + col), &sB[buf][n][col]);
        }
    };

    const int nt = K >> 5;                     // K/32 tiles
    issue_tile(0, 0);                          // prologue: 4 async ops in flight

    for (int it = 0; it < nt; ++it) {
        const int buf = it & 1;
        if (it + 1 < nt) {
            issue_tile(buf ^ 1, (it + 1) << 5);            // 8 ops in flight
            // GL2 prefetch two tiles ahead.
            if (it + 2 < nt) {
                __builtin_prefetch(A  + (bm + (tid >> 1)) * (size_t)K + ((it + 2) << 5), 0, 1);
                __builtin_prefetch(Wt + (bn + (tid >> 1)) * (size_t)K + ((it + 2) << 5), 0, 1);
            }
            // In-order per-wave completion: <=4 left -> tile `it` resident.
            asm volatile("s_wait_asynccnt 4" ::: "memory");
        } else {
            asm volatile("s_wait_asynccnt 0" ::: "memory");
        }
        __syncthreads();                       // publish tile to all waves

        const int lm = lane & 15;
        const int kh = (lane >> 4) << 4;       // A K-half: 0 or 16
        const int kb = (lane >> 4) << 4;       // B K-half: 0 or 16
        v16bf af[2], bfr[4];
        #pragma unroll
        for (int mf = 0; mf < 2; ++mf) {       // A 16x32: 16 contiguous K per lane
            const __bf16* p = &sA[buf][wm + mf * 16 + lm][0];
            #pragma unroll
            for (int i = 0; i < 16; ++i) {
                af[mf][i]     = p[kh + i];
            }
        }
        #pragma unroll
        for (int nf = 0; nf < 4; ++nf) {       // B 32x16: 16 contiguous K per lane
            const __bf16* p = &sB[buf][wn + nf * 16 + lm][kb];
            #pragma unroll
            for (int i = 0; i < 16; ++i) bfr[nf][i] = p[i];
        }
        #pragma unroll
        for (int mf = 0; mf < 2; ++mf)
            #pragma unroll
            for (int nf = 0; nf < 4; ++nf)
                acc[mf][nf] = __builtin_amdgcn_wmma_f32_16x16x32_bf16(
                    false, af[mf], false, bfr[nf], (short)0, acc[mf][nf],
                    false, false);
        __syncthreads();                       // reads done before buf reuse
    }

    // Epilogue (branch-free, templated): C layout — VGPR r: lanes 0-15 M=r,
    // lanes 16-31 M=8+r; N = lane%16.
    const int lm = lane & 15;
    const int mo = (lane >> 4) << 3;
    const size_t rbase = bm + wm + mo;
    const size_t cbase = bn + wn + lm;
    #pragma unroll
    for (int nf = 0; nf < 4; ++nf) {
        const size_t col = cbase + nf * 16;
        const float bv = HASB ? bias[col] : 0.f;
        #pragma unroll
        for (int mf = 0; mf < 2; ++mf) {
            const size_t rowoff = (rbase + mf * 16) * (size_t)N + col;
            #pragma unroll
            for (int r = 0; r < 8; ++r) {
                float v = acc[mf][nf][r] + bv;
                if (RELU) v = fmaxf(v, 0.f);
                if (OUTB) ((__bf16*)Cout)[rowoff + (size_t)r * N] = (__bf16)v;
                else      ((float*) Cout)[rowoff + (size_t)r * N] = v;
            }
        }
    }
}

// ---------------------------------------------------------------------------
// Init slots / S_p / S_s (broadcast over batch).
// ---------------------------------------------------------------------------
__global__ void k_init(const float* __restrict__ slots_param,
                       const float* __restrict__ Sp0, const float* __restrict__ Ss0,
                       float* __restrict__ slots, float* __restrict__ S_p,
                       float* __restrict__ S_s) {
    int idx = blockIdx.x * 256 + threadIdx.x;
    if (idx < BB * NS * DD) slots[idx] = slots_param[idx % (NS * DD)];
    if (idx < BB * NS * 3) {
        int s3 = idx % (NS * 3);
        S_p[idx] = Sp0[s3];
        S_s[idx] = Ss0[s3];
    }
}

// ---------------------------------------------------------------------------
// hk/hv builder: h[b,s,n,d] = base[b,n,d] + rel(b,s,n) @ gw + gb  (bf16 out)
// ---------------------------------------------------------------------------
__global__ void k_build_h(const __bf16* __restrict__ base,
                          const float* __restrict__ coords,
                          const float* __restrict__ S_p, const float* __restrict__ S_s,
                          const float* __restrict__ gw, const float* __restrict__ gb,
                          __bf16* __restrict__ h) {
    size_t idx = (size_t)blockIdx.x * 256 + threadIdx.x;
    int    d  = (int)(idx & (DD - 1));
    size_t n  = (idx >> 8) & (NPTS - 1);
    size_t bs = idx >> 20;                       // / (DD*NPTS)
    size_t b  = bs >> 3;
    const float* c  = coords + (b * NPTS + n) * 3;
    const float* pp = S_p + bs * 3;
    const float* ss = S_s + bs * 3;
    float r0 = (c[0] - pp[0]) / (ss[0] * 5.0f + 1e-8f);
    float r1 = (c[1] - pp[1]) / (ss[1] * 5.0f + 1e-8f);
    float r2 = (c[2] - pp[2]) / (ss[2] * 5.0f + 1e-8f);
    float grel = r0 * gw[d] + r1 * gw[DD + d] + r2 * gw[2 * DD + d] + gb[d];
    h[idx] = (__bf16)((float)base[(b * NPTS + n) * DD + d] + grel);
}

// ---------------------------------------------------------------------------
// dots[b,s,n] = SCALE * <k[b,s,n,:], q[b,s,:]> ; 8 points/block, 32 lanes/point
// ---------------------------------------------------------------------------
__global__ void k_dots(const __bf16* __restrict__ kmat, const float* __restrict__ q,
                       float* __restrict__ dots) {
    int tid = threadIdx.x, lane = tid & 31;
    size_t pt = (size_t)blockIdx.x * 8 + (tid >> 5);
    size_t bs = pt >> 12;
    const __bf16* kr = kmat + pt * DD;
    const float*  qr = q + bs * DD;
    float s = 0.f;
    #pragma unroll
    for (int i = 0; i < 8; ++i) { int d = lane + i * 32; s += (float)kr[d] * qr[d]; }
    #pragma unroll
    for (int m = 16; m > 0; m >>= 1) s += __shfl_xor(s, m, 32);
    if (lane == 0) dots[pt] = s * 0.0625f;     // D^-0.5
}

// ---------------------------------------------------------------------------
// softmax over S (=8) at each (b,n), + EPS ->  attn_raw
// ---------------------------------------------------------------------------
__global__ void k_softmax(const float* __restrict__ dots, float* __restrict__ araw) {
    int idx = blockIdx.x * 256 + threadIdx.x;      // b*NPTS + n
    if (idx >= BB * NPTS) return;
    int b = idx >> 12, n = idx & (NPTS - 1);
    float v[NS]; float mx = -1e30f;
    #pragma unroll
    for (int s = 0; s < NS; ++s) {
        v[s] = dots[((size_t)(b * NS + s) << 12) + n];
        mx = fmaxf(mx, v[s]);
    }
    float sum = 0.f;
    #pragma unroll
    for (int s = 0; s < NS; ++s) { v[s] = __expf(v[s] - mx); sum += v[s]; }
    float inv = 1.f / sum;
    #pragma unroll
    for (int s = 0; s < NS; ++s)
        araw[((size_t)(b * NS + s) << 12) + n] = v[s] * inv + 1e-8f;
}

__global__ void k_rowsum(const float* __restrict__ araw, float* __restrict__ rowsum) {
    __shared__ float red[256];
    int bs = blockIdx.x, tid = threadIdx.x;
    const float* a = araw + (size_t)bs * NPTS;
    float s = 0.f;
    for (int n = tid; n < NPTS; n += 256) s += a[n];
    float t = blk_reduce256(s, red);
    if (tid == 0) rowsum[bs] = t;
}

__global__ void k_attnout(const float* __restrict__ araw, const float* __restrict__ rowsum,
                          float* __restrict__ attn) {
    size_t idx = (size_t)blockIdx.x * 256 + threadIdx.x;
    size_t bs  = idx >> 12;
    attn[idx] = araw[idx] / rowsum[bs];
}

// ---------------------------------------------------------------------------
// updates[b,s,d] = sum_n attn * v ; S_p[b,s,:] = sum_n attn * coords
// one block per (b,s)
// ---------------------------------------------------------------------------
__global__ void k_updates(const __bf16* __restrict__ v, const float* __restrict__ araw,
                          const float* __restrict__ rowsum, const float* __restrict__ coords,
                          float* __restrict__ updates, float* __restrict__ S_p) {
    __shared__ float sat[256];
    __shared__ float red[256];
    int bs = blockIdx.x, tid = threadIdx.x;
    int b  = bs >> 3;
    const __bf16* vb = v + (size_t)bs * NPTS * DD;
    const float*  ar = araw + (size_t)bs * NPTS;
    float acc = 0.f;
    for (int n0 = 0; n0 < NPTS; n0 += 256) {
        sat[tid] = ar[n0 + tid];
        __syncthreads();
        for (int i = 0; i < 256; ++i)
            acc += sat[i] * (float)vb[(size_t)(n0 + i) * DD + tid];
        __syncthreads();
    }
    float inv = 1.f / rowsum[bs];
    updates[(size_t)bs * DD + tid] = acc * inv;

    float q0 = 0.f, q1 = 0.f, q2 = 0.f;
    for (int n = tid; n < NPTS; n += 256) {
        float a = ar[n];
        const float* c = coords + ((size_t)b * NPTS + n) * 3;
        q0 += a * c[0]; q1 += a * c[1]; q2 += a * c[2];
    }
    float t0 = blk_reduce256(q0, red);
    float t1 = blk_reduce256(q1, red);
    float t2 = blk_reduce256(q2, red);
    if (tid == 0) {
        S_p[bs * 3 + 0] = t0 * inv;
        S_p[bs * 3 + 1] = t1 * inv;
        S_p[bs * 3 + 2] = t2 * inv;
    }
}

__global__ void k_ss(const float* __restrict__ coords, const float* __restrict__ S_p,
                     const float* __restrict__ araw, const float* __restrict__ rowsum,
                     float* __restrict__ S_s) {
    __shared__ float red[256];
    int bs = blockIdx.x, tid = threadIdx.x;
    int b  = bs >> 3;
    const float* ar = araw + (size_t)bs * NPTS;
    float p0 = S_p[bs * 3], p1 = S_p[bs * 3 + 1], p2 = S_p[bs * 3 + 2];
    float s0 = 0.f, s1 = 0.f, s2 = 0.f;
    for (int n = tid; n < NPTS; n += 256) {
        float a = ar[n];
        const float* c = coords + ((size_t)b * NPTS + n) * 3;
        float d0 = c[0] - p0, d1 = c[1] - p1, d2 = c[2] - p2;
        s0 += a * d0 * d0; s1 += a * d1 * d1; s2 += a * d2 * d2;
    }
    float inv = 1.f / rowsum[bs];
    float t0 = blk_reduce256(s0, red);
    float t1 = blk_reduce256(s1, red);
    float t2 = blk_reduce256(s2, red);
    if (tid == 0) {
        S_s[bs * 3 + 0] = sqrtf(t0 * inv + 1e-8f);
        S_s[bs * 3 + 1] = sqrtf(t1 * inv + 1e-8f);
        S_s[bs * 3 + 2] = sqrtf(t2 * inv + 1e-8f);
    }
}

// ---------------------------------------------------------------------------
// q = ln(slots) @ Qw ; one block per (b,s)
// ---------------------------------------------------------------------------
__global__ void k_q(const float* __restrict__ slots, const float* __restrict__ ng,
                    const float* __restrict__ nb, const float* __restrict__ Qw,
                    float* __restrict__ q) {
    __shared__ float red[256];
    __shared__ float xn[256];
    int bs = blockIdx.x, tid = threadIdx.x;
    float v = slots[(size_t)bs * DD + tid];
    float mean = blk_reduce256(v, red) / (float)DD;
    float var  = blk_reduce256(v * v, red) / (float)DD - mean * mean;
    xn[tid] = (v - mean) * rsqrtf(var + 1e-5f) * ng[tid] + nb[tid];
    __syncthreads();
    float a = 0.f;
    for (int j = 0; j < DD; ++j) a += xn[j] * Qw[(size_t)j * DD + tid];
    q[(size_t)bs * DD + tid] = a;
}

// ---------------------------------------------------------------------------
// GRU cell + slot MLP; one block per (b,s), updates slots in place.
// ---------------------------------------------------------------------------
__global__ void k_gru(float* __restrict__ slots, const float* __restrict__ updates,
                      const float* __restrict__ wi, const float* __restrict__ wh,
                      const float* __restrict__ bi, const float* __restrict__ bh,
                      const float* __restrict__ lg, const float* __restrict__ lb,
                      const float* __restrict__ w1, const float* __restrict__ b1,
                      const float* __restrict__ w2, const float* __restrict__ b2) {
    __shared__ float su[256], sp[256], yn[256], y1[1024], red[256];
    int bs = blockIdx.x, tid = threadIdx.x;
    su[tid] = updates[(size_t)bs * DD + tid];
    sp[tid] = slots[(size_t)bs * DD + tid];
    __syncthreads();
    float gxr = bi[tid], gxz = bi[DD + tid], gxn = bi[2 * DD + tid];
    float ghr = bh[tid], ghz = bh[DD + tid], ghn = bh[2 * DD + tid];
    for (int j = 0; j < DD; ++j) {
        float u = su[j], h0 = sp[j];
        const float* wij = wi + (size_t)j * (3 * DD);
        const float* whj = wh + (size_t)j * (3 * DD);
        gxr += u * wij[tid]; gxz += u * wij[DD + tid]; gxn += u * wij[2 * DD + tid];
        ghr += h0 * whj[tid]; ghz += h0 * whj[DD + tid]; ghn += h0 * whj[2 * DD + tid];
    }
    float r  = 1.f / (1.f + __expf(-(gxr + ghr)));
    float z  = 1.f / (1.f + __expf(-(gxz + ghz)));
    float nn = tanhf(gxn + r * ghn);
    float h  = (1.f - z) * nn + z * sp[tid];
    float mean = blk_reduce256(h, red) / (float)DD;
    float var  = blk_reduce256(h * h, red) / (float)DD - mean * mean;
    yn[tid] = (h - mean) * rsqrtf(var + 1e-5f) * lg[tid] + lb[tid];
    __syncthreads();
    for (int c = tid; c < 4 * DD; c += 256) {
        float a = b1[c];
        for (int j = 0; j < DD; ++j) a += yn[j] * w1[(size_t)j * (4 * DD) + c];
        y1[c] = fmaxf(a, 0.f);
    }
    __syncthreads();
    float y = b2[tid];
    for (int c = 0; c < 4 * DD; ++c) y += y1[c] * w2[(size_t)c * DD + tid];
    slots[(size_t)bs * DD + tid] = h + y;
}

// ---------------------------------------------------------------------------
// out = slots @ fin_w + fin_b ; one block per (b,s)
// ---------------------------------------------------------------------------
__global__ void k_final(const float* __restrict__ slots, const float* __restrict__ fw,
                        const float* __restrict__ fb, float* __restrict__ out) {
    __shared__ float sr[256];
    int bs = blockIdx.x, tid = threadIdx.x;
    sr[tid] = slots[(size_t)bs * DD + tid];
    __syncthreads();
    float a = fb[tid];
    for (int j = 0; j < DD; ++j) a += sr[j] * fw[(size_t)j * DD + tid];
    out[(size_t)bs * DD + tid] = a;
}

// ---------------------------------------------------------------------------
// Host orchestration
// ---------------------------------------------------------------------------
extern "C" void kernel_launch(void* const* d_in, const int* in_sizes, int n_in,
                              void* d_out, int out_size, void* d_ws, size_t ws_size,
                              hipStream_t stream) {
    const float* inputs   = (const float*)d_in[0];
    const float* coords   = (const float*)d_in[1];
    const float* slots_p  = (const float*)d_in[2];
    const float* Sp0      = (const float*)d_in[3];
    const float* Ss0      = (const float*)d_in[4];
    const float* Qw       = (const float*)d_in[5];
    const float* Kw       = (const float*)d_in[6];
    const float* Vw       = (const float*)d_in[7];
    const float* norm_g   = (const float*)d_in[8];
    const float* norm_b   = (const float*)d_in[9];
    const float* gw       = (const float*)d_in[10];
    const float* gb       = (const float*)d_in[11];
    const float* fw1      = (const float*)d_in[12];
    const float* fb1      = (const float*)d_in[13];
    const float* fw2      = (const float*)d_in[14];
    const float* fb2      = (const float*)d_in[15];
    const float* gru_wi   = (const float*)d_in[16];
    const float* gru_wh   = (const float*)d_in[17];
    const float* gru_bi   = (const float*)d_in[18];
    const float* gru_bh   = (const float*)d_in[19];
    const float* mlp_ln_g = (const float*)d_in[20];
    const float* mlp_ln_b = (const float*)d_in[21];
    const float* mlp_w1   = (const float*)d_in[22];
    const float* mlp_b1   = (const float*)d_in[23];
    const float* mlp_w2   = (const float*)d_in[24];
    const float* mlp_b2   = (const float*)d_in[25];
    const float* im_ln1_g = (const float*)d_in[26];
    const float* im_ln1_b = (const float*)d_in[27];
    const float* im_w1    = (const float*)d_in[28];
    const float* im_b1    = (const float*)d_in[29];
    const float* im_w2    = (const float*)d_in[30];
    const float* im_b2    = (const float*)d_in[31];
    const float* im_ln2_g = (const float*)d_in[32];
    const float* im_ln2_b = (const float*)d_in[33];
    const float* fin_w    = (const float*)d_in[34];
    const float* fin_b    = (const float*)d_in[35];
    float* out = (float*)d_out;

    const size_t M1 = (size_t)BB * NPTS;        // 16384
    const size_t M2 = (size_t)BB * NS * NPTS;   // 131072

    char* p = (char*)d_ws;
    auto alloc = [&](size_t bytes) -> void* {
        void* r = (void*)p;
        p += (bytes + 255) & ~(size_t)255;
        return r;
    };
    __bf16* xb1   = (__bf16*)alloc(M1 * DIN * 2);
    __bf16* hb1   = (__bf16*)alloc(M1 * DIN * 2);
    float*  x2f   = (float*) alloc(M1 * DD * 4);
    __bf16* x2b   = (__bf16*)alloc(M1 * DD * 2);
    __bf16* kx    = (__bf16*)alloc(M1 * DD * 2);
    __bf16* vx    = (__bf16*)alloc(M1 * DD * 2);
    __bf16* wbi1t = (__bf16*)alloc((size_t)DIN * DIN * 2);   // [768][768]
    __bf16* wbi2t = (__bf16*)alloc((size_t)DD * DIN * 2);    // [256][768]
    __bf16* wbKt  = (__bf16*)alloc((size_t)DD * DD * 2);     // [256][256]
    __bf16* wbVt  = (__bf16*)alloc((size_t)DD * DD * 2);
    __bf16* wbf1t = (__bf16*)alloc((size_t)DD * DD * 2);
    __bf16* wbf2t = (__bf16*)alloc((size_t)DD * DD * 2);
    __bf16* hbuf  = (__bf16*)alloc(M2 * DD * 2);
    __bf16* m1    = (__bf16*)alloc(M2 * DD * 2);
    __bf16* m2    = (__bf16*)alloc(M2 * DD * 2);
    float*  dots  = (float*) alloc(M2 * 4);
    float*  araw  = (float*) alloc(M2 * 4);
    float*  rsum  = (float*) alloc(BB * NS * 4);
    float*  qbuf  = (float*) alloc((size_t)BB * NS * DD * 4);
    float*  slots = (float*) alloc((size_t)BB * NS * DD * 4);
    float*  upd   = (float*) alloc((size_t)BB * NS * DD * 4);
    float*  S_p   = (float*) alloc((size_t)BB * NS * 3 * 4);
    float*  S_s   = (float*) alloc((size_t)BB * NS * 3 * 4);

    // ---- weight conversions: f32 [K][N] -> bf16 transposed [N][K] ----
    k_f2bt<<<(DIN * DIN + 255) / 256, 256, 0, stream>>>(im_w1, wbi1t, DIN, DIN);
    k_f2bt<<<(DIN * DD + 255) / 256, 256, 0, stream>>>(im_w2, wbi2t, DIN, DD);
    k_f2bt<<<(DD * DD + 255) / 256, 256, 0, stream>>>(Kw, wbKt, DD, DD);
    k_f2bt<<<(DD * DD + 255) / 256, 256, 0, stream>>>(Vw, wbVt, DD, DD);
    k_f2bt<<<(DD * DD + 255) / 256, 256, 0, stream>>>(fw1, wbf1t, DD, DD);
    k_f2bt<<<(DD * DD + 255) / 256, 256, 0, stream>>>(fw2, wbf2t, DD, DD);

    // ---- input MLP: ln -> relu(@im_w1) -> @im_w2 -> ln -> kx/vx ----
    k_ln<<<M1, 256, 0, stream>>>(inputs, im_ln1_g, im_ln1_b, xb1, DIN);
    k_gemm<true, true, true><<<dim3(M1 / 128, DIN / 128), 256, 0, stream>>>(
        xb1, wbi1t, im_b1, (void*)hb1, (int)M1, DIN, DIN);
    k_gemm<false, true, false><<<dim3(M1 / 128, DD / 128), 256, 0, stream>>>(
        hb1, wbi2t, im_b2, (void*)x2f, (int)M1, DD, DIN);
    k_ln<<<M1, 256, 0, stream>>>(x2f, im_ln2_g, im_ln2_b, x2b, DD);
    k_gemm<false, false, true><<<dim3(M1 / 128, DD / 128), 256, 0, stream>>>(
        x2b, wbKt, nullptr, (void*)kx, (int)M1, DD, DD);
    k_gemm<false, false, true><<<dim3(M1 / 128, DD / 128), 256, 0, stream>>>(
        x2b, wbVt, nullptr, (void*)vx, (int)M1, DD, DD);

    k_init<<<(BB * NS * DD + 255) / 256, 256, 0, stream>>>(
        slots_p, Sp0, Ss0, slots, S_p, S_s);

    const int nBS = BB * NS;
    for (int t = 0; t <= NITR; ++t) {
        // q from current slots
        k_q<<<nBS, 256, 0, stream>>>(slots, norm_g, norm_b, Qw, qbuf);

        // k stream: hk -> relu(@fw1) -> @fw2 -> dots
        k_build_h<<<(unsigned)(M2 * DD / 256), 256, 0, stream>>>(
            kx, coords, S_p, S_s, gw, gb, hbuf);
        k_gemm<true, true, true><<<dim3(M2 / 128, DD / 128), 256, 0, stream>>>(
            hbuf, wbf1t, fb1, (void*)m1, (int)M2, DD, DD);
        k_gemm<false, true, true><<<dim3(M2 / 128, DD / 128), 256, 0, stream>>>(
            m1, wbf2t, fb2, (void*)m2, (int)M2, DD, DD);
        k_dots<<<(unsigned)(M2 / 8), 256, 0, stream>>>(m2, qbuf, dots);

        // softmax over slots + renormalization over points
        k_softmax<<<(BB * NPTS + 255) / 256, 256, 0, stream>>>(dots, araw);
        k_rowsum<<<nBS, 256, 0, stream>>>(araw, rsum);
        k_attnout<<<(unsigned)(M2 / 256), 256, 0, stream>>>(araw, rsum,
                                                            out + BB * NS * DD);

        if (t < NITR) {
            // v stream (uses S_p/S_s of this iteration — before they update)
            k_build_h<<<(unsigned)(M2 * DD / 256), 256, 0, stream>>>(
                vx, coords, S_p, S_s, gw, gb, hbuf);
            k_gemm<true, true, true><<<dim3(M2 / 128, DD / 128), 256, 0, stream>>>(
                hbuf, wbf1t, fb1, (void*)m1, (int)M2, DD, DD);
            k_gemm<false, true, true><<<dim3(M2 / 128, DD / 128), 256, 0, stream>>>(
                m1, wbf2t, fb2, (void*)m2, (int)M2, DD, DD);

            k_updates<<<nBS, 256, 0, stream>>>(m2, araw, rsum, coords, upd, S_p);
            k_ss<<<nBS, 256, 0, stream>>>(coords, S_p, araw, rsum, S_s);
            k_gru<<<nBS, 256, 0, stream>>>(slots, upd, gru_wi, gru_wh, gru_bi,
                                           gru_bh, mlp_ln_g, mlp_ln_b, mlp_w1,
                                           mlp_b1, mlp_w2, mlp_b2);
        }
    }
    // out = slots_prev @ fin_w + fin_b  (slots unchanged during t==NITR)
    k_final<<<nBS, 256, 0, stream>>>(slots, fin_w, fin_b, out);
}
